// GCN2LayerConcat_26560077758924
// MI455X (gfx1250) — matrix operand order repaired and checked by reading, compiled, not run on hardware
//
#include <hip/hip_runtime.h>

// Flip to 0 if the async-to-LDS inline asm is rejected by the assembler.
#define USE_ASYNC_LDS 1

#define N_NODES 65536
#define N_GRAPHS 32
#define NODES_PER_GRAPH 2048
#define N_EDGES 1048576
#define F_IN 16
#define H1 64
#define H2 64
#define OUT_DIM 12

typedef __attribute__((ext_vector_type(16))) _Float16 v16h;
typedef __attribute__((ext_vector_type(8)))  float    v8f;

#if USE_ASYNC_LDS
typedef __attribute__((address_space(3))) char lds_char_t;

__device__ inline void async_lds_copy16(void* lds, const void* gmem) {
    const lds_char_t* lp = (const lds_char_t*)lds;   // 32-bit LDS pointer (1 VGPR)
    asm volatile("global_load_async_to_lds_b128 %0, %1, off"
                 :: "v"(lp), "v"(gmem) : "memory");
}
__device__ inline void async_lds_wait() {
    asm volatile("s_wait_asynccnt 0x0" ::: "memory");
}
#endif

// Cooperative copy of launch-constant packed weights into LDS (16B/thread/iter).
__device__ inline void stage_weights(_Float16* sdst, const _Float16* gsrc, int nbytes) {
    for (int off = (int)threadIdx.x * 16; off < nbytes; off += (int)blockDim.x * 16) {
#if USE_ASYNC_LDS
        async_lds_copy16((char*)sdst + off, (const char*)gsrc + off);
#else
        *(float4*)((char*)sdst + off) = *(const float4*)((const char*)gsrc + off);
#endif
    }
#if USE_ASYNC_LDS
    async_lds_wait();
#endif
    __syncthreads();
}

// ---------------------------------------------------------------------------
// Degree / normalization:  deg[i] = 1 (self loop) + #incoming edges,
// then dinv[i] = rsqrt(deg[i])  (deg >= 1 always).
// ---------------------------------------------------------------------------
__global__ void deg_init_kernel(float* deg) {
    int i = blockIdx.x * blockDim.x + threadIdx.x;
    if (i < N_NODES) deg[i] = 1.0f;
}

__global__ void deg_accum_kernel(const int* __restrict__ dst, float* deg) {
    int e = blockIdx.x * blockDim.x + threadIdx.x;
    if (e < N_EDGES) atomicAdd(&deg[dst[e]], 1.0f);
}

__global__ void dinv_kernel(float* deg_dinv) {
    int i = blockIdx.x * blockDim.x + threadIdx.x;
    if (i < N_NODES) deg_dinv[i] = rsqrtf(deg_dinv[i]);
}

// ---------------------------------------------------------------------------
// Weight packing: rewrite W (KxN f32, row-major) into the exact per-lane
// WMMA B-operand register image (f16), one 32-byte record per (tile, lane):
//   B layout (16-bit B 32x16): lane L -> N = L&15, hg = L>>4; j<8 -> K = hg*8+j, else K = 16+hg*8+(j-8)
// pack[(tile*32 + lane)*16 + j]
// ---------------------------------------------------------------------------
__global__ void pack_w1_kernel(const float* __restrict__ W1, _Float16* __restrict__ pack1) {
    int i = blockIdx.x * blockDim.x + threadIdx.x;   // 4 tiles * 32 lanes * 16 = 2048
    if (i >= 4 * 32 * 16) return;
    int j    = i & 15;
    int lane = (i >> 4) & 31;
    int nt   = i >> 9;
    int hg   = lane >> 4;
    int col  = nt * 16 + (lane & 15);
    int k    = (j < 8) ? (hg * 8 + j) : (16 + hg * 8 + (j - 8));   // K padded 16 -> 32
    pack1[i] = (k < F_IN) ? (_Float16)W1[k * H1 + col] : (_Float16)0.0f;
}

__global__ void pack_w2_kernel(const float* __restrict__ W2, _Float16* __restrict__ pack2) {
    int i = blockIdx.x * blockDim.x + threadIdx.x;   // 8 tiles * 32 lanes * 16 = 4096
    if (i >= 8 * 32 * 16) return;
    int j    = i & 15;
    int lane = (i >> 4) & 31;
    int t    = i >> 9;                               // t = nt*2 + s  (s = K-step)
    int s    = t & 1;
    int nt   = t >> 1;
    int hg   = lane >> 4;
    int col  = nt * 16 + (lane & 15);
    int k    = s * 32 + ((j < 8) ? (hg * 8 + j) : (16 + hg * 8 + (j - 8)));
    pack2[i] = (_Float16)W2[k * H2 + col];
}

// ---------------------------------------------------------------------------
// GEMM 1:  h = x @ W1   (65536x16 @ 16x64), K zero-padded to 32.
// 256-thread blocks = 8 waves; packed B staged once into LDS and shared.
// A layout (16-bit A 16x32): lane -> M = L&15, hg = L>>4;
//   element j<8 : K = hg*8 + j ; element j>=8 : K = 16 + hg*8 + (j-8) (padded)
// C/D: VGPR r -> M = r (lanes 0-15) / r+8 (lanes 16-31), N = L&15.
// ---------------------------------------------------------------------------
__global__ __launch_bounds__(256) void gemm1_wmma_kernel(
        const float* __restrict__ x, const _Float16* __restrict__ pack1,
        float* __restrict__ h) {
    __shared__ _Float16 sB[4 * 32 * 16];             // 4 KB
    stage_weights(sB, pack1, 4 * 32 * 16 * 2);

    const int lane = threadIdx.x & 31;
    const int wave = threadIdx.x >> 5;
    const int m  = lane & 15;
    const int hg = lane >> 4;
    const int nodeBase = (blockIdx.x * 8 + wave) * 16;

    const float* xrow = x + (size_t)(nodeBase + m) * F_IN + hg * 8;
    v16h a;
#pragma unroll
    for (int j = 0; j < 8; ++j)  a[j] = (_Float16)xrow[j];
#pragma unroll
    for (int j = 8; j < 16; ++j) a[j] = (_Float16)0.0f;

    const v16h* Bs = (const v16h*)sB;
#pragma unroll
    for (int nt = 0; nt < 4; ++nt) {
        v16h b = Bs[nt * 32 + lane];
        v8f c = {};
        c = __builtin_amdgcn_wmma_f32_16x16x32_f16(false, a, false, b,
                                                   (short)0, c, false, false);
        const int col = nt * 16 + m;
        float* outp = h + (size_t)(nodeBase + hg * 8) * H1 + col;
#pragma unroll
        for (int r = 0; r < 8; ++r) outp[(size_t)r * H1] = c[r];
    }
}

// ---------------------------------------------------------------------------
// GEMM 2:  h = hin @ W2   (65536x64 @ 64x64).  K = 64 -> two chained WMMAs.
// ---------------------------------------------------------------------------
__global__ __launch_bounds__(256) void gemm2_wmma_kernel(
        const float* __restrict__ hin, const _Float16* __restrict__ pack2,
        float* __restrict__ hout) {
    __shared__ _Float16 sB[8 * 32 * 16];             // 8 KB
    stage_weights(sB, pack2, 8 * 32 * 16 * 2);

    const int lane = threadIdx.x & 31;
    const int wave = threadIdx.x >> 5;
    const int m  = lane & 15;
    const int hg = lane >> 4;
    const int nodeBase = (blockIdx.x * 8 + wave) * 16;

    const float* row = hin + (size_t)(nodeBase + m) * H1;
    v16h a[2];
#pragma unroll
    for (int s = 0; s < 2; ++s) {
        const float* rp = row + s * 32 + hg * 8;
#pragma unroll
        for (int j = 0; j < 8; ++j) a[s][j]     = (_Float16)rp[j];
#pragma unroll
        for (int j = 0; j < 8; ++j) a[s][j + 8] = (_Float16)rp[16 + j];
    }

    const v16h* Bs = (const v16h*)sB;
#pragma unroll
    for (int nt = 0; nt < 4; ++nt) {
        v8f c = {};
#pragma unroll
        for (int s = 0; s < 2; ++s) {
            v16h b = Bs[(nt * 2 + s) * 32 + lane];
            c = __builtin_amdgcn_wmma_f32_16x16x32_f16(false, a[s], false, b,
                                                       (short)0, c, false, false);
        }
        const int col = nt * 16 + m;
        float* outp = hout + (size_t)(nodeBase + hg * 8) * H1 + col;
#pragma unroll
        for (int r = 0; r < 8; ++r) outp[(size_t)r * H1] = c[r];
    }
}

// ---------------------------------------------------------------------------
// acc[i][c] = h[i][c] * dinv[i]^2 + b[c]    (self-loop term + bias)
// ---------------------------------------------------------------------------
__global__ void init_acc_kernel(const float* __restrict__ h,
                                const float* __restrict__ dinv,
                                const float* __restrict__ bias,
                                float* __restrict__ acc) {
    int i = blockIdx.x * blockDim.x + threadIdx.x;   // over N_NODES*64
    if (i >= N_NODES * 64) return;
    int node = i >> 6;
    int c = i & 63;
    float di = dinv[node];
    acc[i] = h[i] * di * di + bias[c];
}

// ---------------------------------------------------------------------------
// Edge scatter:  acc[dst] += h[src] * dinv[src]*dinv[dst]
// 16 lanes per edge, float4 per lane -> 256B coalesced row burst, L2-resident
// (h is 16 MB << 192 MB L2). src/dst/dinv loads are shared by the half-wave.
// ---------------------------------------------------------------------------
__global__ void edge_scatter_kernel(const int* __restrict__ src,
                                    const int* __restrict__ dst,
                                    const float* __restrict__ dinv,
                                    const float* __restrict__ h,
                                    float* __restrict__ acc) {
    unsigned tid = blockIdx.x * blockDim.x + threadIdx.x;   // N_EDGES*16 threads
    unsigned e = tid >> 4;
    if (e >= N_EDGES) return;
    int c = (tid & 15) * 4;
    int s = src[e];
    int d = dst[e];
    float norm = dinv[s] * dinv[d];
    const float4 hv = *(const float4*)(h + (size_t)s * 64 + c);
    float* ap = acc + (size_t)d * 64 + c;
    atomicAdd(ap + 0, hv.x * norm);
    atomicAdd(ap + 1, hv.y * norm);
    atomicAdd(ap + 2, hv.z * norm);
    atomicAdd(ap + 3, hv.w * norm);
}

__global__ void tanh_kernel(float* buf) {
    int i = blockIdx.x * blockDim.x + threadIdx.x;
    if (i < N_NODES * 64) buf[i] = tanhf(buf[i]);
}

// ---------------------------------------------------------------------------
// Final FC:  out[g,o] = sum_k h2[g*131072 + k] * Wfc[k,o] + bfc[o]
// Wfc is 6.3 MB -> L2-resident across all 32 graphs.
// ---------------------------------------------------------------------------
__global__ void fc_init_kernel(const float* __restrict__ bfc, float* out) {
    int i = blockIdx.x * blockDim.x + threadIdx.x;
    if (i < N_GRAPHS * OUT_DIM) out[i] = bfc[i % OUT_DIM];
}

__global__ void fc_kernel(const float* __restrict__ h2,
                          const float* __restrict__ Wfc,
                          float* __restrict__ out) {
    __shared__ float sred[OUT_DIM];
    const int g = blockIdx.x;
    const int K = NODES_PER_GRAPH * H2;          // 131072
    const int CHUNK = K / gridDim.y;
    const int k0 = blockIdx.y * CHUNK;

    const float* hg = h2 + (size_t)g * K;
    float acc[OUT_DIM];
#pragma unroll
    for (int o = 0; o < OUT_DIM; ++o) acc[o] = 0.0f;

    for (int k = k0 + threadIdx.x; k < k0 + CHUNK; k += blockDim.x) {
        float hv = hg[k];
        const float* wrow = Wfc + (size_t)k * OUT_DIM;
#pragma unroll
        for (int o = 0; o < OUT_DIM; ++o) acc[o] += hv * wrow[o];
    }

    if (threadIdx.x < OUT_DIM) sred[threadIdx.x] = 0.0f;
    __syncthreads();
#pragma unroll
    for (int o = 0; o < OUT_DIM; ++o) atomicAdd(&sred[o], acc[o]);
    __syncthreads();
    if (threadIdx.x < OUT_DIM)
        atomicAdd(&out[g * OUT_DIM + threadIdx.x], sred[threadIdx.x]);
}

// ---------------------------------------------------------------------------
// Launcher.  Input order (setup_inputs): x, edge_index, batch, W1, b1, W2, b2,
// Wfc, bfc.  Output: [32, 12] float32.
// Workspace (floats): dinv[65536] | bufA[65536*64] | bufB[65536*64]
//                     | pack1[2048 f16] | pack2[4096 f16]
// ---------------------------------------------------------------------------
extern "C" void kernel_launch(void* const* d_in, const int* in_sizes, int n_in,
                              void* d_out, int out_size, void* d_ws, size_t ws_size,
                              hipStream_t stream) {
    (void)in_sizes; (void)n_in; (void)out_size; (void)ws_size;

    const float* x    = (const float*)d_in[0];
    const int*   ei   = (const int*)d_in[1];
    const int*   src  = ei;
    const int*   dst  = ei + N_EDGES;
    const float* W1   = (const float*)d_in[3];
    const float* b1   = (const float*)d_in[4];
    const float* W2   = (const float*)d_in[5];
    const float* b2   = (const float*)d_in[6];
    const float* Wfc  = (const float*)d_in[7];
    const float* bfc  = (const float*)d_in[8];
    float* out        = (float*)d_out;

    float* dinv = (float*)d_ws;
    float* bufA = dinv + N_NODES;
    float* bufB = bufA + (size_t)N_NODES * 64;
    _Float16* pack1 = (_Float16*)(bufB + (size_t)N_NODES * 64);  // 16B/32B aligned
    _Float16* pack2 = pack1 + 4 * 32 * 16;

    // --- weight packing (launch-constant, tiny) ---
    pack_w1_kernel<<<8, 256, 0, stream>>>(W1, pack1);
    pack_w2_kernel<<<16, 256, 0, stream>>>(W2, pack2);

    // --- degree / dinv ---
    deg_init_kernel<<<N_NODES / 256, 256, 0, stream>>>(dinv);
    deg_accum_kernel<<<N_EDGES / 256, 256, 0, stream>>>(dst, dinv);
    dinv_kernel<<<N_NODES / 256, 256, 0, stream>>>(dinv);

    const int gemmBlocks = N_NODES / (16 * 8);         // 512
    const int featBlocks = (N_NODES * 64) / 256;       // 16384
    const int edgeBlocks = (N_EDGES * 16) / 256;       // 65536

    // --- layer 1 ---
    gemm1_wmma_kernel<<<gemmBlocks, 256, 0, stream>>>(x, pack1, bufA);
    init_acc_kernel<<<featBlocks, 256, 0, stream>>>(bufA, dinv, b1, bufB);
    edge_scatter_kernel<<<edgeBlocks, 256, 0, stream>>>(src, dst, dinv, bufA, bufB);
    tanh_kernel<<<featBlocks, 256, 0, stream>>>(bufB);

    // --- layer 2 ---
    gemm2_wmma_kernel<<<gemmBlocks, 256, 0, stream>>>(bufB, pack2, bufA);
    init_acc_kernel<<<featBlocks, 256, 0, stream>>>(bufA, dinv, b2, bufB);
    edge_scatter_kernel<<<edgeBlocks, 256, 0, stream>>>(src, dst, dinv, bufA, bufB);
    tanh_kernel<<<featBlocks, 256, 0, stream>>>(bufB);

    // --- readout FC ---
    fc_init_kernel<<<2, 256, 0, stream>>>(bfc, out);
    fc_kernel<<<dim3(N_GRAPHS, 64), 256, 0, stream>>>(bufB, Wfc, out);
}
